// DeepStreamOutput_17214228922495
// MI455X (gfx1250) — compile-verified, matrix operand order checked by reading, not run on hardware
//
#include <hip/hip_runtime.h>
#include <hip/hip_bf16.h>

typedef __attribute__((ext_vector_type(16))) _Float16 v16h;
typedef __attribute__((ext_vector_type(8)))  float    v8f;

#define NB     8
#define NA     25200
#define NCV    80
#define NMV    32
#define ROW    117        // 5 + 80 + 32
#define MHW    25600      // 160*160
#define MAXDET 100

// d_out layout (flat concat of the 4 outputs, in return order)
#define OFF_BOXES   0
#define OFF_SCORES  (NB * MAXDET * 4)             // 3200
#define OFF_CLASSES (OFF_SCORES + NB * MAXDET)    // 4000
#define OFF_MASKS   (OFF_CLASSES + NB * MAXDET)   // 4800

__device__ __forceinline__ float fast_sigmoid(float x) {
    // v_exp_f32 + v_rcp_f32, no IEEE-divide sequence
    return __builtin_amdgcn_rcpf(1.0f + __expf(-x));
}

// ---------------------------------------------------------------------------
// Kernel 1: per-(batch,slot) scalars. Only the gathered 100 rows do real work;
// everything else writes zeros (reference zero-pads unmatched slots).
// ---------------------------------------------------------------------------
__global__ void det_scalars(const float* __restrict__ preds,
                            const int* __restrict__ sel,
                            float* __restrict__ out) {
    int t = blockIdx.x * blockDim.x + threadIdx.x;
    if (t >= NB * MAXDET) return;
    int b = t / MAXDET;
    int n = t % MAXDET;
    int bi = sel[3 * n + 0];
    int bx = sel[3 * n + 2];

    float x0 = 0.f, y0 = 0.f, x1 = 0.f, y1 = 0.f, sc = 0.f, cl = 0.f;
    if (bi == b) {
        const float* r = preds + ((size_t)bi * NA + bx) * ROW;
        float cx = r[0], cy = r[1], w = r[2], h = r[3], obj = r[4];
        float m = r[5];
        int mi = 0;
        for (int c = 1; c < NCV; ++c) {
            float v = r[5 + c];
            if (v > m) { m = v; mi = c; }      // strict '>' == jnp.argmax first-max
        }
        // cxcywh @ convert_matrix -> xyxy
        x0 = cx - 0.5f * w;  y0 = cy - 0.5f * h;
        x1 = cx + 0.5f * w;  y1 = cy + 0.5f * h;
        sc = m * obj;
        cl = (float)mi;
    }
    float* ob = out + OFF_BOXES + (size_t)t * 4;
    ob[0] = x0; ob[1] = y0; ob[2] = x1; ob[3] = y1;
    out[OFF_SCORES  + t] = sc;
    out[OFF_CLASSES + t] = cl;
}

// ---------------------------------------------------------------------------
// Kernel 2: zero-fill the UNMATCHED mask planes only (matched planes are fully
// written by the WMMA kernel -> every byte written exactly once).
// 25600 % 128 == 0, so each wave's float4 span lies in one slot: uniform skip.
// ---------------------------------------------------------------------------
__global__ void fill_masks(float4* __restrict__ masks4,
                           const int* __restrict__ sel) {
    long i = (long)blockIdx.x * blockDim.x + threadIdx.x;
    if (i >= (long)NB * MAXDET * MHW / 4) return;
    long slot = (i * 4) / MHW;
    int b = (int)(slot / MAXDET);
    int n = (int)(slot % MAXDET);
    if (sel[3 * n] == b) return;               // matched: WMMA kernel writes it
    masks4[i] = make_float4(0.f, 0.f, 0.f, 0.f);
}

// ---------------------------------------------------------------------------
// Kernel 3: masks = sigmoid(coef[n,0:32] . proto[n,0:32,p]) via
// v_wmma_f32_16x16x32_f16 (K=32 == NM). A = coef broadcast over M rows
// (documented 16-bit A 16x32 layout: lanes 0-15 -> K{0..7,16..23},
// lanes 16-31 -> K{8..15,24..31}); B = proto tile, pixels along N
// (mirrored layout), so D columns are pixels.
// Two tiles per iteration: lanes 0-15 store tile0's c[0], lanes 16-31 store
// tile1's c[0] -> one fully-coalesced 128B store per tile pair, no branch.
// ---------------------------------------------------------------------------
__global__ void __launch_bounds__(256) mask_wmma(
        const float* __restrict__ preds,
        const int*   __restrict__ sel,
        const float* __restrict__ pooled,
        float*       __restrict__ out_masks) {
    const int n    = blockIdx.y;               // detection 0..99
    const int bi   = sel[3 * n + 0];
    const int bx   = sel[3 * n + 2];
    const int lane = threadIdx.x & 31;
    const int wave = threadIdx.x >> 5;
    const int hi   = lane >> 4;                // half-wave selects K subset
    const int pl   = lane & 15;                // pixel within 16-wide tile

    // A operand: mask coefficients, broadcast over all 16 M rows.
    const float* crow = preds + ((size_t)bi * NA + bx) * ROW + 5 + NCV;
    v16h a;
#pragma unroll
    for (int e = 0; e < 8; ++e) a[e]     = (_Float16)crow[hi * 8 + e];        // K 0..7 / 8..15
#pragma unroll
    for (int e = 0; e < 8; ++e) a[8 + e] = (_Float16)crow[16 + hi * 8 + e];   // K 16..23 / 24..31

    const float* pb = pooled + (size_t)n * NMV * MHW;
    float* ob = out_masks + ((size_t)bi * MAXDET + n) * MHW;

    const int wgl = blockIdx.x * 8 + wave;     // 0..199 waves per detection
#pragma unroll 1
    for (int t = 0; t < 4; ++t) {              // 4 iterations x 2 tiles = 8 tiles
        const int tile0 = wgl * 8 + t * 2;     // even tile of the pair
        const int p0    = tile0 * 16 + pl;
        const int p1    = p0 + 16;

        // prefetch next pair's stream (global_prefetch_b8)
        __builtin_prefetch(pb + (size_t)(hi * 8) * MHW + p0 + 32, 0, 1);

        v16h b0, b1;
#pragma unroll
        for (int e = 0; e < 8; ++e) {
            const size_t r0 = (size_t)(hi * 8 + e) * MHW;
            const size_t r1 = (size_t)(16 + hi * 8 + e) * MHW;
            b0[e]     = (_Float16)pb[r0 + p0];
            b0[8 + e] = (_Float16)pb[r1 + p0];
            b1[e]     = (_Float16)pb[r0 + p1];
            b1[8 + e] = (_Float16)pb[r1 + p1];
        }

        v8f c0 = {}, c1 = {};
        c0 = __builtin_amdgcn_wmma_f32_16x16x32_f16(
                false, a, false, b0, (short)0, c0, false, false);
        c1 = __builtin_amdgcn_wmma_f32_16x16x32_f16(
                false, a, false, b1, (short)0, c1, false, false);

        // lanes 0-15: pixel tile0*16+lane (from c0); lanes 16-31: tile1 (c1)
        float x = (lane < 16) ? c0[0] : c1[0];
        ob[tile0 * 16 + lane] = fast_sigmoid(x);
    }
}

// ---------------------------------------------------------------------------
extern "C" void kernel_launch(void* const* d_in, const int* in_sizes, int n_in,
                              void* d_out, int out_size, void* d_ws, size_t ws_size,
                              hipStream_t stream) {
    const float* preds  = (const float*)d_in[0];
    // d_in[1] (protos) only contributes shape info in the reference -> unused
    const int*   sel    = (const int*)d_in[2];
    const float* pooled = (const float*)d_in[3];
    float* out = (float*)d_out;

    det_scalars<<<dim3((NB * MAXDET + 255) / 256), dim3(256), 0, stream>>>(preds, sel, out);

    // 8*100*25600/4 = 5,120,000 float4 -> 20,000 blocks of 256
    fill_masks<<<dim3(20000), dim3(256), 0, stream>>>((float4*)(out + OFF_MASKS), sel);

    // 100 detections x 25 blocks x 8 waves x 8 tiles x 16 px = 25600 px/det
    mask_wmma<<<dim3(25, MAXDET), dim3(256), 0, stream>>>(preds, sel, pooled, out + OFF_MASKS);
}
